// STFT_7473243095228
// MI455X (gfx1250) — matrix operand order repaired and checked by reading, not compiled, not run on hardware
//
#include <hip/hip_runtime.h>
#include <hip/hip_bf16.h>

typedef __attribute__((ext_vector_type(16))) __bf16 v16bf;
typedef __attribute__((ext_vector_type(8)))  float  v8f;

#define NSIG     1048576
#define FILT     1024
#define HOP      512
#define CUTOFF   513
#define NFRAMES  2049
#define NROWS    1026

__device__ __forceinline__ v8f wmma_bf16(v16bf a, v16bf b, v8f c) {
  // (neg_a, A, neg_b, B, c_mod, C, reuse_a, reuse_b)
  return __builtin_amdgcn_wmma_f32_16x16x32_bf16(false, a, false, b, (short)0, c, false, false);
}

__device__ __forceinline__ void ld8(const float* __restrict__ p, float* v) {
  const float4* q = reinterpret_cast<const float4*>(p);
  float4 a = q[0], b = q[1];
  v[0] = a.x; v[1] = a.y; v[2] = a.z; v[3] = a.w;
  v[4] = b.x; v[5] = b.y; v[6] = b.z; v[7] = b.w;
}

// f32 -> bf16 hi + bf16 lo (residual). hi*hi + lo*hi + hi*lo accumulated in
// f32 gives ~16 mantissa bits of product precision (near-fp32 result).
// Lowers to v_cvt_pk_bf16_f32 + v_fma_mix_f32_bf16 (verified in round-1 asm).
__device__ __forceinline__ void split16(const float* v, v16bf& hi, v16bf& lo) {
#pragma unroll
  for (int e = 0; e < 16; ++e) {
    float x = v[e];
    __bf16 h = (__bf16)x;
    hi[e] = h;
    lo[e] = (__bf16)(x - (float)h);
  }
}

// Grid: x = t-blocks (256 frames), y = freq-tile pairs, z = batch.
// Block = 256 threads = 8 waves; wave (wf,wt): wf selects the freq tile, wt
// its group of 4 t-tiles. Each wave owns the REAL tile (rows R..R+15) and the
// IMAG tile (rows R+513..): both share B fragments, and the magnitude
// epilogue runs on in-register accumulators (no giant intermediate).
__global__ __launch_bounds__(256)
void stft_mag_kernel(const float* __restrict__ sig_all,
                     const float* __restrict__ basis,
                     float* __restrict__ out_all) {
  const int tid   = threadIdx.x;
  const int wave  = tid >> 5;
  const int lane  = tid & 31;
  const int wf    = wave & 1;
  const int wt    = wave >> 1;
  const int m     = lane & 15;   // A-tile row / B-tile frame
  const int khalf = lane >> 4;

  const int R     = (blockIdx.y * 2 + wf) * 16;      // freq-tile row base
  const int Tbase = blockIdx.x * 256 + wt * 64;      // wave's first frame
  const int b     = blockIdx.z;

  if (R > 512) return;        // wave-uniform: no valid output rows
  if (Tbase > 2048) return;   // wave-uniform: no valid frames

  const float* __restrict__ sig = sig_all + (size_t)b * NSIG;
  float* __restrict__ out = out_all + (size_t)b * (size_t)(CUTOFF * NFRAMES);

  // A fragment (contiguous k order, same as B): elems 0..15 = K kbaseA..+15,
  // kbaseA = 16*(lane>=16).
  const int kbaseA = khalf * 16;
  const int rRe = R + m;                                 // <= 527 < 1026
  const int rIm = min(R + CUTOFF + m, NROWS - 1);        // clamp pad rows
  const float* __restrict__ aRe = basis + (size_t)rRe * FILT + kbaseA;
  const float* __restrict__ aIm = basis + (size_t)rIm * FILT + kbaseA;

  // B fragment (32x16): 16 contiguous K values starting at 16*(lane>=16).
  // Hoisted per-tile bases: g = base + k0 (+e).
  const int kbaseB = khalf * 16;
  const int gb0 = (Tbase +  0 + m) * HOP + kbaseB - HOP;
  const int gb1 = (Tbase + 16 + m) * HOP + kbaseB - HOP;
  const int gb2 = (Tbase + 32 + m) * HOP + kbaseB - HOP;
  const int gb3 = (Tbase + 48 + m) * HOP + kbaseB - HOP;
  const int lo0 = (Tbase +  0) * HOP - HOP;              // wave-uniform bounds
  const int lo1 = (Tbase + 16) * HOP - HOP;
  const int lo2 = (Tbase + 32) * HOP - HOP;
  const int lo3 = (Tbase + 48) * HOP - HOP;

  // Named accumulators (arrays caused whole-file v_mov_b64 rotation in RA).
  v8f accRe0 = {}, accRe1 = {}, accRe2 = {}, accRe3 = {};
  v8f accIm0 = {}, accIm1 = {}, accIm2 = {}, accIm3 = {};

  for (int k0 = 0; k0 < FILT; k0 += 32) {
    float va[16];
    v16bf aReHi, aReLo, aImHi, aImLo;
    ld8(aRe + k0,      va);
    ld8(aRe + k0 + 8,  va + 8);
    split16(va, aReHi, aReLo);
    ld8(aIm + k0,      va);
    ld8(aIm + k0 + 8,  va + 8);
    split16(va, aImHi, aImLo);

#define DO_TILE(GB, LO, ACCRE, ACCIM)                                          \
    {                                                                          \
      float vb[16];                                                            \
      const int gbase = (GB) + k0;                                             \
      if ((LO) + k0 >= 0 && (LO) + 15 * HOP + k0 + 31 < NSIG) {                \
        ld8(sig + gbase,     vb);      /* fast path: contiguous b128 loads */  \
        ld8(sig + gbase + 8, vb + 8);                                          \
      } else {                                                                 \
        _Pragma("unroll")                                                      \
        for (int e = 0; e < 16; ++e) { /* np.pad 'reflect' indexing */         \
          int g = gbase + e;                                                   \
          g = (g < 0) ? -g : g;                                                \
          g = (g >= NSIG) ? (2 * NSIG - 2 - g) : g;                            \
          vb[e] = sig[g];                                                      \
        }                                                                      \
      }                                                                        \
      v16bf bHi, bLo;                                                          \
      split16(vb, bHi, bLo);                                                   \
      ACCRE = wmma_bf16(aReHi, bHi, ACCRE);                                    \
      ACCRE = wmma_bf16(aReLo, bHi, ACCRE);                                    \
      ACCRE = wmma_bf16(aReHi, bLo, ACCRE);                                    \
      ACCIM = wmma_bf16(aImHi, bHi, ACCIM);                                    \
      ACCIM = wmma_bf16(aImLo, bHi, ACCIM);                                    \
      ACCIM = wmma_bf16(aImHi, bLo, ACCIM);                                    \
    }

    DO_TILE(gb0, lo0, accRe0, accIm0)
    DO_TILE(gb1, lo1, accRe1, accIm1)
    DO_TILE(gb2, lo2, accRe2, accIm2)
    DO_TILE(gb3, lo3, accRe3, accIm3)
#undef DO_TILE
  }

  // Epilogue: magnitude from paired accumulators.
  // C/D layout: elem j = row R+j (+8 for lanes 16..31), lane&15 = frame.
#define STORE_TILE(I, ACCRE, ACCIM)                                            \
  {                                                                            \
    const int t = Tbase + 16 * (I) + m;                                        \
    if (t < NFRAMES) {                                                         \
      _Pragma("unroll")                                                        \
      for (int j = 0; j < 8; ++j) {                                            \
        const int f = R + j + 8 * khalf;                                       \
        if (f < CUTOFF) {                                                      \
          float re = ACCRE[j];                                                 \
          float im = ACCIM[j];                                                 \
          out[(size_t)f * NFRAMES + t] = sqrtf(re * re + im * im);             \
        }                                                                      \
      }                                                                        \
    }                                                                          \
  }

  STORE_TILE(0, accRe0, accIm0)
  STORE_TILE(1, accRe1, accIm1)
  STORE_TILE(2, accRe2, accIm2)
  STORE_TILE(3, accRe3, accIm3)
#undef STORE_TILE
}

extern "C" void kernel_launch(void* const* d_in, const int* in_sizes, int n_in,
                              void* d_out, int out_size, void* d_ws, size_t ws_size,
                              hipStream_t stream) {
  (void)in_sizes; (void)n_in; (void)out_size; (void)d_ws; (void)ws_size;
  const float* input = (const float*)d_in[0];   // (16, 1048576) f32
  const float* basis = (const float*)d_in[1];   // (1026, 1024) f32
  float* out = (float*)d_out;                   // (16, 513, 2049) f32
  dim3 grid(9, 17, 16);   // ceil(2049/256) t-blocks, ceil(33 freq-tiles / 2), 16 batches
  dim3 block(256);
  hipLaunchKernelGGL(stft_mag_kernel, grid, block, 0, stream, input, basis, out);
}
